// DOL_31164282700395
// MI455X (gfx1250) — matrix operand order repaired and checked by reading, not compiled, not run on hardware
//
#include <hip/hip_runtime.h>
#include <math.h>

typedef __attribute__((ext_vector_type(16))) __bf16 v16bf;
typedef __attribute__((ext_vector_type(8)))  float  v8f;
typedef __attribute__((ext_vector_type(4)))  unsigned int v4u;
typedef __attribute__((ext_vector_type(8)))  int v8i;
typedef __attribute__((ext_vector_type(4)))  int v4i;
typedef unsigned short u16;
typedef unsigned int   u32;

#define HAVE_TDM 0
#if defined(__has_builtin)
#if __has_builtin(__builtin_amdgcn_tensor_load_to_lds) && __has_builtin(__builtin_amdgcn_s_wait_tensorcnt)
#undef HAVE_TDM
#define HAVE_TDM 1
#endif
#endif

// ---------------- bf16 helpers (manual, RNE) ----------------
__device__ __forceinline__ u16 f2bf(float x) {
  u32 u = __float_as_uint(x);
  u32 r = u + 0x7FFFu + ((u >> 16) & 1u);
  return (u16)(r >> 16);
}
__device__ __forceinline__ float bf2f(u16 h) {
  return __uint_as_float(((u32)h) << 16);
}

union FragU { v16bf v; uint4 q[2]; };

// A fragment (16x32 bf16): lane holds row m=(lane&15); K = {h*8..h*8+7} U {16+h*8..16+h*8+7}
__device__ __forceinline__ v16bf load_frag_a(const u16* rowbase, int lhalf) {
  FragU f;
  f.q[0] = *(const uint4*)(rowbase + lhalf * 8);
  f.q[1] = *(const uint4*)(rowbase + 16 + lhalf * 8);
  return f.v;
}
// B fragment from LDS stored transposed [n][32k]: lane holds col n=(lane&15); K = {h*8..h*8+7} U {16+h*8..16+h*8+7}
__device__ __forceinline__ v16bf load_frag_b(const u16* rowbase, int lhalf) {
  FragU f;
  f.q[0] = *(const uint4*)(rowbase + lhalf * 8);
  f.q[1] = *(const uint4*)(rowbase + 16 + lhalf * 8);
  return f.v;
}
__device__ __forceinline__ v8f wmma_bf16(v16bf a, v16bf b, v8f c) {
  return __builtin_amdgcn_wmma_f32_16x16x32_bf16(false, a, false, b, (short)0, c, false, false);
}

// ================= start conv + per-node adapter =================
// x0[b,c,n,t] = (sw[c]*s + sb[c]) + W2[n] * relu(s*alpha[n] + beta[n]) + b2[n]
// where s = input[b,t,n]; alpha = W1*sw, beta = W1*sb + b1 (per node).
__global__ __launch_bounds__(128) void k_start(
    const float* __restrict__ in, const float* __restrict__ sw, const float* __restrict__ sb,
    const float* __restrict__ w1, const float* __restrict__ b1,
    const float* __restrict__ w2, const float* __restrict__ b2,
    float* __restrict__ x0)
{
  __shared__ float sW1[16*32];
  __shared__ float sW2[32*16];
  __shared__ float sAl[16], sBe[16], sB2[32], sSW[32], sSB[32];
  const int n = blockIdx.x;
  const int tid = threadIdx.x;
  for (int e = tid; e < 512; e += 128) sW1[e] = w1[(size_t)n*512 + e];
  for (int e = tid; e < 512; e += 128) sW2[e] = w2[(size_t)n*512 + e];
  if (tid < 32) { sB2[tid] = b2[n*32 + tid]; sSW[tid] = sw[tid]; sSB[tid] = sb[tid]; }
  __syncthreads();
  if (tid < 16) {
    float al = 0.f, be = b1[n*16 + tid];
    for (int c = 0; c < 32; ++c) { al += sW1[tid*32+c]*sSW[c]; be += sW1[tid*32+c]*sSB[c]; }
    sAl[tid] = al; sBe[tid] = be;
  }
  __syncthreads();
  for (int col = tid; col < 32*13; col += 128) {
    int b = col / 13, t = col % 13;
    float s = in[((size_t)b*13 + t)*1024 + n];
    float h[16];
    #pragma unroll
    for (int d = 0; d < 16; ++d) { float v = s*sAl[d] + sBe[d]; h[d] = v > 0.f ? v : 0.f; }
    for (int c = 0; c < 32; ++c) {
      float v = sSW[c]*s + sSB[c] + sB2[c];
      #pragma unroll
      for (int d = 0; d < 16; ++d) v += sW2[c*16+d]*h[d];
      x0[(((size_t)b*32 + c)*1024 + n)*13 + t] = v;
    }
  }
}

// ================= adaptive adjacency: softmax(relu(nv1@nv2)) row-wise, stored TRANSPOSED bf16 =================
__global__ __launch_bounds__(256) void k_adp(
    const float* __restrict__ nv1, const float* __restrict__ nv2, u16* __restrict__ adpT)
{
  __shared__ float z[1024];
  __shared__ float red[256];
  const int v = blockIdx.x, tid = threadIdx.x;
  float nv[10];
  #pragma unroll
  for (int e = 0; e < 10; ++e) nv[e] = nv1[v*10 + e];
  float lmax = -1e30f;
  for (int w = tid; w < 1024; w += 256) {
    float a = 0.f;
    #pragma unroll
    for (int e = 0; e < 10; ++e) a += nv[e]*nv2[e*1024 + w];
    a = a > 0.f ? a : 0.f;
    z[w] = a;
    lmax = fmaxf(lmax, a);
  }
  red[tid] = lmax; __syncthreads();
  for (int s = 128; s > 0; s >>= 1) { if (tid < s) red[tid] = fmaxf(red[tid], red[tid+s]); __syncthreads(); }
  float m = red[0]; __syncthreads();
  float lsum = 0.f;
  for (int w = tid; w < 1024; w += 256) { float e = __expf(z[w]-m); z[w] = e; lsum += e; }
  red[tid] = lsum; __syncthreads();
  for (int s = 128; s > 0; s >>= 1) { if (tid < s) red[tid] += red[tid+s]; __syncthreads(); }
  float inv = 1.f / red[0];
  for (int w = tid; w < 1024; w += 256) adpT[(size_t)w*1024 + v] = f2bf(z[w]*inv);
}

// ================= transpose + convert support matrix: outT[w][v] = bf16(in[v][w]) =================
__global__ __launch_bounds__(256) void k_cvt_t(const float* __restrict__ in, u16* __restrict__ outT)
{
  __shared__ u16 tile[32][33];
  const int bx = blockIdx.x * 32, by = blockIdx.y * 32;
  const int tx = threadIdx.x & 31, ty = threadIdx.x >> 5;
  for (int r = ty; r < 32; r += 8)
    tile[r][tx] = f2bf(in[(size_t)(by + r)*1024 + bx + tx]);
  __syncthreads();
  for (int r = ty; r < 32; r += 8)
    outT[(size_t)(bx + r)*1024 + by + tx] = tile[tx][r];
}

// ================= dilated gated conv: gxb[b,c,t',n] = tanh(filt)*sigmoid(gate), bf16 [M=(b,c,t'),N] =================
__global__ __launch_bounds__(128) void k_gate(
    const float* __restrict__ x, u16* __restrict__ gxb,
    const float* __restrict__ fw, const float* __restrict__ fb,
    const float* __restrict__ gw, const float* __restrict__ gb,
    int d, int Tin, int Tout)
{
  __shared__ float xs[32][16];
  __shared__ float sfw[2048], sgw[2048];
  __shared__ float sfb[32], sgb[32];
  const int n = blockIdx.x, b = blockIdx.y, tid = threadIdx.x;
  for (int e = tid; e < 2048; e += 128) { sfw[e] = fw[e]; sgw[e] = gw[e]; }
  if (tid < 32) { sfb[tid] = fb[tid]; sgb[tid] = gb[tid]; }
  for (int e = tid; e < 32*Tin; e += 128) {
    int c = e / Tin, t = e % Tin;
    xs[c][t] = x[(((size_t)b*32 + c)*1024 + n)*Tin + t];
  }
  __syncthreads();
  for (int idx = tid; idx < 32*Tout; idx += 128) {
    int co = idx / Tout, t = idx % Tout;
    float f = sfb[co], g = sgb[co];
    for (int ci = 0; ci < 32; ++ci) {
      float a = xs[ci][t], bb = xs[ci][t+d];
      const float* fwp = &sfw[(co*32+ci)*2];
      const float* gwp = &sgw[(co*32+ci)*2];
      f += fwp[0]*a + fwp[1]*bb;
      g += gwp[0]*a + gwp[1]*bb;
    }
    float val = tanhf(f) * (1.f/(1.f+__expf(-g)));
    gxb[(((size_t)b*32 + co)*Tout + t)*1024 + n] = f2bf(val);
  }
}

// ================= node-diffusion GEMM: O[M,1024] = X[M,1024] * A[1024,1024] (A given TRANSPOSED) =================
// block tile 128x128, 8 waves as 2(M)x4(N), wave tile 64x32 (4x2 WMMA accums)
// A tile staged via Tensor Data Mover (TDM) when available; B tile staged with contiguous b128 stores.
__global__ __launch_bounds__(256) void k_gemm_node(
    const u16* __restrict__ X, const u16* __restrict__ At, u16* __restrict__ O, int M)
{
  __shared__ __align__(16) u16 lA[128*32];
  __shared__ __align__(16) u16 lB[128*32];   // [n][k]
  const int tid = threadIdx.x;
  const int lane = tid & 31, wid = tid >> 5;
  const int lrow = lane & 15, lhalf = lane >> 4;
  const int n0 = blockIdx.x * 128;
  const int m0 = blockIdx.y * 128;
  const int waveM = wid >> 2, waveN = wid & 3;
  const v8f vz = {0.f,0.f,0.f,0.f,0.f,0.f,0.f,0.f};
  v8f acc[4][2];
  #pragma unroll
  for (int i = 0; i < 4; ++i) { acc[i][0] = vz; acc[i][1] = vz; }

  const int sr = tid >> 1, sc = (tid & 1) * 16;   // staging: 128 rows x 32 cols, 2 x b128 per thread

  for (int k0 = 0; k0 < 1024; k0 += 32) {
#if HAVE_TDM
    if (wid == 0) {
      // D# per CDNA5 ISA ch.8: 2D tile 32(dim0) x 128(dim1), 2B elements, row stride 1024 elems.
      unsigned long long ga = (unsigned long long)(size_t)X + (((size_t)m0*1024 + k0) * 2ull);
      v4u g0;
      g0.x = 1u;                                              // count=1 (valid user descriptor)
      g0.y = (u32)(size_t)lA;                                 // lds_addr (bytes)
      g0.z = (u32)ga;                                         // global_addr[31:0]
      g0.w = (u32)((ga >> 32) & 0x01FFFFFFu) | (2u << 30);    // global_addr[56:32] | type=2
      v8i g1;
      g1[0] = 0x00010000;                                     // data_size=2B; no multicast/pad/iterate
      g1[1] = (int)(1024u << 16);                             // tensor_dim0 = 1024 (lo16)
      g1[2] = (int)(((u32)M & 0xFFFFu) << 16);                // tensor_dim1 lo16
      g1[3] = (int)((((u32)M >> 16) & 0xFFFFu) | (32u << 16)); // tensor_dim1 hi16 | tile_dim0=32
      g1[4] = 128;                                            // tile_dim1=128, tile_dim2=0
      g1[5] = 1024;                                            // tensor_dim0_stride lo32
      g1[6] = 0;
      g1[7] = 0;
      v4i gz4 = {0, 0, 0, 0};
      v8i gz8 = {0, 0, 0, 0, 0, 0, 0, 0};
      __builtin_amdgcn_tensor_load_to_lds(g0, g1, gz4, gz4, gz8, 0);
    }
#else
    {
      const uint4* as = (const uint4*)(X + ((size_t)(m0 + sr))*1024 + k0 + sc);
      uint4 a0 = as[0], a1 = as[1];
      *(uint4*)&lA[sr*32 + sc]     = a0;
      *(uint4*)&lA[sr*32 + sc + 8] = a1;
    }
    if (k0 + 32 < 1024)
      __builtin_prefetch(X + ((size_t)(m0 + sr))*1024 + k0 + 32 + sc, 0, 1);
#endif
    // stage B tile [128 n][32 k] from pre-transposed support (contiguous rows)
    {
      const uint4* bs = (const uint4*)(At + ((size_t)(n0 + sr))*1024 + k0 + sc);
      uint4 b0 = bs[0], b1 = bs[1];
      *(uint4*)&lB[sr*32 + sc]     = b0;
      *(uint4*)&lB[sr*32 + sc + 8] = b1;
    }
    if (k0 + 32 < 1024)
      __builtin_prefetch(At + ((size_t)(n0 + sr))*1024 + k0 + 32 + sc, 0, 1);
#if HAVE_TDM
    __builtin_amdgcn_s_wait_tensorcnt(0);
#endif
    __syncthreads();
    v16bf bf0 = load_frag_b(&lB[(waveN*32 + 0*16 + lrow)*32], lhalf);
    v16bf bf1 = load_frag_b(&lB[(waveN*32 + 1*16 + lrow)*32], lhalf);
    #pragma unroll
    for (int mf = 0; mf < 4; ++mf) {
      v16bf af = load_frag_a(&lA[(waveM*64 + mf*16 + lrow)*32], lhalf);
      acc[mf][0] = wmma_bf16(af, bf0, acc[mf][0]);
      acc[mf][1] = wmma_bf16(af, bf1, acc[mf][1]);
    }
    __syncthreads();
  }
  #pragma unroll
  for (int mf = 0; mf < 4; ++mf) {
    int mb = m0 + waveM*64 + mf*16 + lhalf*8;
    #pragma unroll
    for (int nf = 0; nf < 2; ++nf) {
      int nn = n0 + waveN*32 + nf*16 + lrow;
      #pragma unroll
      for (int i = 0; i < 8; ++i)
        O[((size_t)(mb + i))*1024 + nn] = f2bf(acc[mf][nf][i]);
    }
  }
}

// ================= GCN combine (one 32-wide K group per pass), accumulate into xn[b,co,n,t] =================
// mode: 0 = write, 1 = accumulate, 2 = accumulate + bias + residual (final group)
__global__ __launch_bounds__(256) void k_gemm_gcn(
    const float* __restrict__ W, int g, const u16* __restrict__ H,
    const float* __restrict__ bias, const float* __restrict__ resid,
    float* __restrict__ xn, int Tin, int Tout, int mode)
{
  __shared__ __align__(16) u16 lA[32*32];
  __shared__ __align__(16) u16 lB[128*32];
  const int tid = threadIdx.x;
  const int lane = tid & 31, wid = tid >> 5;
  const int lrow = lane & 15, lhalf = lane >> 4;
  const int n0 = blockIdx.x * 128;
  const int t = blockIdx.y;
  const int b = blockIdx.z;
  const v8f vz = {0.f,0.f,0.f,0.f,0.f,0.f,0.f,0.f};
  // stage A: W[:, g*32 : g*32+32] -> bf16 [32][32]
  {
    int e = tid * 4;
    #pragma unroll
    for (int j = 0; j < 4; ++j) {
      int ee = e + j;
      int co = ee >> 5, c = ee & 31;
      lA[ee] = f2bf(W[co*224 + g*32 + c]);
    }
  }
  // stage B transposed: rows c=0..31, cols n0..n0+127 of H[b,c,t,:]
  {
    int c = tid >> 3, nch = (tid & 7) * 16;
    const uint4* bs = (const uint4*)(H + ((((size_t)b*32 + c)*Tout + t)*1024) + n0 + nch);
    uint4 b0 = bs[0], b1 = bs[1];
    const u16* p0 = (const u16*)&b0; const u16* p1 = (const u16*)&b1;
    #pragma unroll
    for (int j = 0; j < 8; ++j) lB[(nch + j)*32 + c] = p0[j];
    #pragma unroll
    for (int j = 0; j < 8; ++j) lB[(nch + 8 + j)*32 + c] = p1[j];
  }
  __syncthreads();
  v16bf bf = load_frag_b(&lB[(wid*16 + lrow)*32], lhalf);
  v16bf a0 = load_frag_a(&lA[(0  + lrow)*32], lhalf);
  v16bf a1 = load_frag_a(&lA[(16 + lrow)*32], lhalf);
  v8f acc0 = wmma_bf16(a0, bf, vz);
  v8f acc1 = wmma_bf16(a1, bf, vz);
  int nn = n0 + wid*16 + lrow;
  #pragma unroll
  for (int mf = 0; mf < 2; ++mf) {
    v8f acc = mf ? acc1 : acc0;
    #pragma unroll
    for (int i = 0; i < 8; ++i) {
      int co = mf*16 + lhalf*8 + i;
      size_t idx = (((size_t)b*32 + co)*1024 + nn)*Tout + t;
      float v = acc[i];
      if (mode > 0) v += xn[idx];
      if (mode == 2)
        v += bias[co] + resid[(((size_t)b*32 + co)*1024 + nn)*Tin + (t + Tin - Tout)];
      xn[idx] = v;
    }
  }
}

// ================= skip conv at the final time step only: skip[b,sc,n] (+)= Ws*gx[b,:,T-1,n] + sb =================
// block tile 256(M) x 64(N); 8 waves each 32x64 (2x4 accums); single K step (K=32)
__global__ __launch_bounds__(256) void k_gemm_skip(
    const float* __restrict__ Ws, const float* __restrict__ sb,
    const u16* __restrict__ gxb, float* __restrict__ skip,
    int Tout, int init)
{
  __shared__ __align__(16) u16 lA[256*32];
  __shared__ __align__(16) u16 lB[64*32];
  const int tid = threadIdx.x;
  const int lane = tid & 31, wid = tid >> 5;
  const int lrow = lane & 15, lhalf = lane >> 4;
  const int j0 = blockIdx.x * 64;
  const int b = j0 >> 10;
  const int n0 = j0 & 1023;
  const v8f vz = {0.f,0.f,0.f,0.f,0.f,0.f,0.f,0.f};
  for (int e = tid; e < 8192; e += 256) lA[e] = f2bf(Ws[e]);
  {
    int c = tid >> 3, nch = (tid & 7) * 8;
    const uint4* bs = (const uint4*)(gxb + ((((size_t)b*32 + c)*Tout + (Tout-1))*1024) + n0 + nch);
    uint4 b0 = bs[0];
    const u16* p0 = (const u16*)&b0;
    #pragma unroll
    for (int j = 0; j < 8; ++j) lB[(nch + j)*32 + c] = p0[j];
  }
  __syncthreads();
  v16bf bf[4];
  #pragma unroll
  for (int nf = 0; nf < 4; ++nf) bf[nf] = load_frag_b(&lB[(nf*16 + lrow)*32], lhalf);
  v8f acc[2][4];
  #pragma unroll
  for (int mf = 0; mf < 2; ++mf) {
    v16bf af = load_frag_a(&lA[(wid*32 + mf*16 + lrow)*32], lhalf);
    #pragma unroll
    for (int nf = 0; nf < 4; ++nf) acc[mf][nf] = wmma_bf16(af, bf[nf], vz);
  }
  #pragma unroll
  for (int mf = 0; mf < 2; ++mf) {
    #pragma unroll
    for (int nf = 0; nf < 4; ++nf) {
      int nn = n0 + nf*16 + lrow;
      #pragma unroll
      for (int i = 0; i < 8; ++i) {
        int sc = wid*32 + mf*16 + lhalf*8 + i;
        size_t idx = ((size_t)b*256 + sc)*1024 + nn;
        float v = acc[mf][nf][i] + sb[sc];
        if (!init) v += skip[idx];
        skip[idx] = v;
      }
    }
  }
}

// ================= end conv: endo[b,ec,n] = relu(Wend * relu(skip) + bend), bf16 out =================
// block tile 128x128, 8 waves 2(M)x4(N), wave tile 64x32; K=256 in 8 steps
__global__ __launch_bounds__(256) void k_gemm_end(
    const float* __restrict__ Wend, const float* __restrict__ bend,
    const float* __restrict__ skip, u16* __restrict__ endo)
{
  __shared__ __align__(16) u16 lA[128*32];
  __shared__ __align__(16) u16 lB[128*32];
  const int tid = threadIdx.x;
  const int lane = tid & 31, wid = tid >> 5;
  const int lrow = lane & 15, lhalf = lane >> 4;
  const int j0 = blockIdx.x * 128;
  const int b = j0 >> 10, n0 = j0 & 1023;
  const int m0 = blockIdx.y * 128;
  const int waveM = wid >> 2, waveN = wid & 3;
  const v8f vz = {0.f,0.f,0.f,0.f,0.f,0.f,0.f,0.f};
  v8f acc[4][2];
  #pragma unroll
  for (int i = 0; i < 4; ++i) { acc[i][0] = vz; acc[i][1] = vz; }
  const int ar = tid >> 1, ac = (tid & 1) * 16;
  const int bk = tid >> 3, bn = (tid & 7) * 16;
  for (int k0 = 0; k0 < 256; k0 += 32) {
    #pragma unroll
    for (int j = 0; j < 16; ++j)
      lA[ar*32 + ac + j] = f2bf(Wend[(size_t)(m0 + ar)*256 + k0 + ac + j]);
    #pragma unroll
    for (int j = 0; j < 16; ++j) {
      float v = skip[((size_t)b*256 + k0 + bk)*1024 + n0 + bn + j];
      lB[(bn + j)*32 + bk] = f2bf(v > 0.f ? v : 0.f);
    }
    __syncthreads();
    v16bf bf0 = load_frag_b(&lB[(waveN*32 + 0*16 + lrow)*32], lhalf);
    v16bf bf1 = load_frag_b(&lB[(waveN*32 + 1*16 + lrow)*32], lhalf);
    #pragma unroll
    for (int mf = 0; mf < 4; ++mf) {
      v16bf af = load_frag_a(&lA[(waveM*64 + mf*16 + lrow)*32], lhalf);
      acc[mf][0] = wmma_bf16(af, bf0, acc[mf][0]);
      acc[mf][1] = wmma_bf16(af, bf1, acc[mf][1]);
    }
    __syncthreads();
  }
  #pragma unroll
  for (int mf = 0; mf < 4; ++mf) {
    #pragma unroll
    for (int nf = 0; nf < 2; ++nf) {
      int nn = n0 + waveN*32 + nf*16 + lrow;
      #pragma unroll
      for (int i = 0; i < 8; ++i) {
        int ec = m0 + waveM*64 + mf*16 + lhalf*8 + i;
        float v = acc[mf][nf][i] + bend[ec];
        endo[((size_t)b*512 + ec)*1024 + nn] = f2bf(v > 0.f ? v : 0.f);
      }
    }
  }
}

// ================= BatchNorm (training-mode batch stats) =================
__global__ __launch_bounds__(256) void k_bn_stats(
    const float* __restrict__ x, int T, const float* __restrict__ g, float* __restrict__ stat)
{
  __shared__ float rs[256], rs2[256];
  const int c = blockIdx.x, tid = threadIdx.x;
  float s = 0.f, s2 = 0.f;
  const int L = 1024*T;
  for (int b = 0; b < 32; ++b) {
    const float* p = x + (((size_t)b*32 + c)*1024)*T;
    for (int i = tid; i < L; i += 256) { float v = p[i]; s += v; s2 += v*v; }
  }
  rs[tid] = s; rs2[tid] = s2; __syncthreads();
  for (int st = 128; st > 0; st >>= 1) {
    if (tid < st) { rs[tid] += rs[tid+st]; rs2[tid] += rs2[tid+st]; }
    __syncthreads();
  }
  if (tid == 0) {
    float cnt = 32.f * (float)L;
    float m = rs[0]/cnt;
    float var = rs2[0]/cnt - m*m;
    stat[c] = m;
    stat[32 + c] = g[c] * rsqrtf(var + 1e-5f);
  }
}

__global__ void k_bn_apply(float* __restrict__ x, int T,
    const float* __restrict__ stat, const float* __restrict__ beta)
{
  size_t i = (size_t)blockIdx.x*blockDim.x + threadIdx.x;
  size_t total = (size_t)32*32*1024*T;
  if (i >= total) return;
  int c = (int)((i / ((size_t)1024*T)) & 31);
  x[i] = (x[i] - stat[c]) * stat[32 + c] + beta[c];
}

// ================= decoder 512 -> 12 =================
__global__ __launch_bounds__(256) void k_dec(
    const u16* __restrict__ endo, const float* __restrict__ dw,
    const float* __restrict__ db, float* __restrict__ out)
{
  int i = blockIdx.x*256 + threadIdx.x;     // 32*12*1024 = 393216 exact
  int b = i / (12*1024);
  int r = i - b*12*1024;
  int p = r >> 10;
  int n = r & 1023;
  float acc = db[p];
  const float* w = dw + p*512;
  const u16* e = endo + (size_t)b*512*1024 + n;
  for (int k = 0; k < 512; ++k) acc += w[k] * bf2f(e[(size_t)k*1024]);
  out[i] = acc;
}

// ================= host orchestration =================
extern "C" void kernel_launch(void* const* d_in, const int* in_sizes, int n_in,
                              void* d_out, int out_size, void* d_ws, size_t ws_size,
                              hipStream_t stream)
{
  (void)in_sizes; (void)n_in; (void)out_size; (void)ws_size;
  const float* in      = (const float*)d_in[0];
  const float* sup0    = (const float*)d_in[1];
  const float* sup1    = (const float*)d_in[2];
  const float* start_w = (const float*)d_in[3];
  const float* start_b = (const float*)d_in[4];
  const float* ad_w1   = (const float*)d_in[5];
  const float* ad_b1   = (const float*)d_in[6];
  const float* ad_w2   = (const float*)d_in[7];
  const float* ad_b2   = (const float*)d_in[8];
  const float* nv1     = (const float*)d_in[9];
  const float* nv2     = (const float*)d_in[10];
  const float* filt_w  = (const float*)d_in[11];
  const float* filt_b  = (const float*)d_in[12];
  const float* gate_w  = (const float*)d_in[13];
  const float* gate_b  = (const float*)d_in[14];
  const float* skip_w  = (const float*)d_in[15];
  const float* skip_b  = (const float*)d_in[16];
  const float* gcn_w   = (const float*)d_in[17];
  const float* gcn_b   = (const float*)d_in[18];
  const float* bn_g    = (const float*)d_in[19];
  const float* bn_b    = (const float*)d_in[20];
  const float* end_w   = (const float*)d_in[21];
  const float* end_b   = (const float*)d_in[22];
  const float* dec_w   = (const float*)d_in[23];
  const float* dec_b   = (const float*)d_in[24];

  char* ws = (char*)d_ws;
  size_t off = 0;
  auto alloc = [&](size_t bytes) -> void* {
    void* p = ws + off; off += (bytes + 255) & ~(size_t)255; return p;
  };
  float* xA   = (float*)alloc((size_t)32*32*1024*13*4);
  float* xB   = (float*)alloc((size_t)32*32*1024*13*4);
  u16*   gxb  = (u16*)  alloc((size_t)32*32*12*1024*2);
  u16*   nb1  = (u16*)  alloc((size_t)32*32*12*1024*2);
  u16*   nb2  = (u16*)  alloc((size_t)32*32*12*1024*2);
  u16*   s0t  = (u16*)  alloc((size_t)1024*1024*2);
  u16*   s1t  = (u16*)  alloc((size_t)1024*1024*2);
  u16*   adpt = (u16*)  alloc((size_t)1024*1024*2);
  float* skp  = (float*)alloc((size_t)32*256*1024*4);
  u16*   endo = (u16*)  alloc((size_t)32*512*1024*2);
  float* stat = (float*)alloc(64*4);

  k_start<<<1024, 128, 0, stream>>>(in, start_w, start_b, ad_w1, ad_b1, ad_w2, ad_b2, xA);
  k_adp<<<1024, 256, 0, stream>>>(nv1, nv2, adpt);
  k_cvt_t<<<dim3(32,32), 256, 0, stream>>>(sup0, s0t);
  k_cvt_t<<<dim3(32,32), 256, 0, stream>>>(sup1, s1t);

  const int dil[8] = {1,2,1,2,1,2,1,2};
  const u16* sups[3] = {s0t, s1t, adpt};
  int Tin = 13;
  float* xc = xA; float* xn = xB;
  for (int i = 0; i < 8; ++i) {
    const int d = dil[i];
    const int Tout = Tin - d;
    const int M = 1024 * Tout;                 // B*C*Tout = 32*32*Tout
    k_gate<<<dim3(1024,32), 128, 0, stream>>>(xc, gxb,
        filt_w + (size_t)i*2048, filt_b + i*32,
        gate_w + (size_t)i*2048, gate_b + i*32, d, Tin, Tout);
    k_gemm_skip<<<512, 256, 0, stream>>>(skip_w + (size_t)i*8192, skip_b + (size_t)i*256,
        gxb, skp, Tout, (i == 0) ? 1 : 0);
    const float* Wl = gcn_w + (size_t)i*32*224;
    dim3 ggrid(8, Tout, 32);
    k_gemm_gcn<<<ggrid, 256, 0, stream>>>(Wl, 0, gxb, gcn_b + i*32, xc, xn, Tin, Tout, 0);
    for (int s = 0; s < 3; ++s) {
      k_gemm_node<<<dim3(8, M/128), 256, 0, stream>>>(gxb, sups[s], nb1, M);
      k_gemm_gcn<<<ggrid, 256, 0, stream>>>(Wl, 2*s+1, nb1, gcn_b + i*32, xc, xn, Tin, Tout, 1);
      k_gemm_node<<<dim3(8, M/128), 256, 0, stream>>>(nb1, sups[s], nb2, M);
      k_gemm_gcn<<<ggrid, 256, 0, stream>>>(Wl, 2*s+2, nb2, gcn_b + i*32, xc, xn, Tin, Tout,
                                            (s == 2) ? 2 : 1);
    }
    k_bn_stats<<<32, 256, 0, stream>>>(xn, Tout, bn_g + i*32, stat);
    size_t total = (size_t)32*32*1024*Tout;
    k_bn_apply<<<(unsigned)((total + 255)/256), 256, 0, stream>>>(xn, Tout, stat, bn_b + i*32);
    float* tmp = xc; xc = xn; xn = tmp;
    Tin = Tout;
  }
  k_gemm_end<<<dim3(256, 4), 256, 0, stream>>>(end_w, end_b, skp, endo);
  k_dec<<<1536, 256, 0, stream>>>(endo, dec_w, dec_b, (float*)d_out);
}
